// Video_encoder_41764261986531
// MI455X (gfx1250) — matrix operand order repaired and checked by reading, not compiled, not run on hardware
//
#include <hip/hip_runtime.h>
#include <hip/hip_bf16.h>
#include <math.h>

// ---------------------------------------------------------------------------
// MI455X (gfx1250) implementation: all GEMM-shaped work on v_wmma_f32_16x16x32_bf16
// (bf16 inputs, fp32 accumulate), everything stateful/pointwise in fp32.
// ---------------------------------------------------------------------------

typedef __attribute__((ext_vector_type(16))) __bf16 v16bf;
typedef __attribute__((ext_vector_type(8)))  __bf16 v8bf;
typedef __attribute__((ext_vector_type(8)))  float  v8f;

constexpr int Bb   = 128;
constexpr int Tt   = 64;
constexpr int Dd   = 2048;
constexpr int Hh   = 512;
constexpr int G3   = 3 * Hh;      // 1536
constexpr int ROUT = 2 * Hh;      // 1024
constexpr int Kc   = 512;         // conv out channels
constexpr int FIN  = 5632;
constexpr int FOUT = 2048;

#define DEV __device__ __forceinline__

DEV v8f zero_v8f() {
  v8f z;
#pragma unroll
  for (int i = 0; i < 8; ++i) z[i] = 0.0f;
  return z;
}

DEV v16bf zero_v16bf() {
  v16bf z;
#pragma unroll
  for (int i = 0; i < 16; ++i) z[i] = (__bf16)0.0f;
  return z;
}

// A fragment (16x32 bf16, row-major A[M][K]): rowbase = &A[m0 + (lane&15)][k0]
// half 0 covers K {0..7, 16..23}, half 1 covers K {8..15, 24..31}  (ISA 7.12.2)
DEV v16bf load_a_frag(const __bf16* rowbase, int hl) {
  v8bf lo = *(const v8bf*)(rowbase + 8 * hl);
  v8bf hi = *(const v8bf*)(rowbase + 16 + 8 * hl);
  return __builtin_shufflevector(lo, hi, 0,1,2,3,4,5,6,7,8,9,10,11,12,13,14,15);
}

// B fragment (32x16 bf16, B[k][n] = W[n][k], W row-major (N x K)):
// rowbase = &W[n0 + (lane&15)][k0]; half 0 -> K {0..7, 16..23}, half 1 -> K {8..15, 24..31}
DEV v16bf load_b_frag(const __bf16* rowbase, int hl) {
  v8bf lo = *(const v8bf*)(rowbase + 8 * hl);
  v8bf hi = *(const v8bf*)(rowbase + 16 + 8 * hl);
  return __builtin_shufflevector(lo, hi, 0,1,2,3,4,5,6,7,8,9,10,11,12,13,14,15);
}

DEV v8f wmma_bf16(v16bf a, v16bf b, v8f c) {
  return __builtin_amdgcn_wmma_f32_16x16x32_bf16(false, a, false, b, (short)0, c,
                                                 false, false);
}

// ---------------------------------------------------------------------------
// Generic GEMM:  C[M][N] = A[M][K] (bf16) * W[N][K]^T (bf16) + bias[N]
// block = 256 threads (8 waves), block tile 64M x 256N, wave tile 32M x 64N.
// Requires M%64==0, N%256==0, K%32==0 (true for all call sites).
// ---------------------------------------------------------------------------
__global__ void gemm_bf16_wmma(const __bf16* __restrict__ A, int lda,
                               const __bf16* __restrict__ W, int ldb,
                               const float* __restrict__ bias,
                               float* __restrict__ C, int ldc, int K) {
  const int lane = threadIdx.x & 31;
  const int wid  = threadIdx.x >> 5;
  const int wm = wid >> 2, wn = wid & 3;
  const int m0 = blockIdx.y * 64 + wm * 32;
  const int n0 = blockIdx.x * 256 + wn * 64;
  const int hl  = (lane >> 4) & 1;
  const int l15 = lane & 15;

  v8f acc[2][4];
#pragma unroll
  for (int mi = 0; mi < 2; ++mi)
#pragma unroll
    for (int ni = 0; ni < 4; ++ni) acc[mi][ni] = zero_v8f();

  const __bf16* pa0 = A + (size_t)(m0 + l15) * lda;
  const __bf16* pa1 = pa0 + (size_t)16 * lda;

  for (int k0 = 0; k0 < K; k0 += 32) {
    v16bf a0 = load_a_frag(pa0 + k0, hl);
    v16bf a1 = load_a_frag(pa1 + k0, hl);
    v16bf bfr[4];
#pragma unroll
    for (int ni = 0; ni < 4; ++ni) {
      const __bf16* pb = W + (size_t)(n0 + 16 * ni + l15) * ldb + k0;
      bfr[ni] = load_b_frag(pb, hl);
    }
#pragma unroll
    for (int ni = 0; ni < 4; ++ni) {
      acc[0][ni] = wmma_bf16(a0, bfr[ni], acc[0][ni]);
      acc[1][ni] = wmma_bf16(a1, bfr[ni], acc[1][ni]);
    }
  }

#pragma unroll
  for (int mi = 0; mi < 2; ++mi)
#pragma unroll
    for (int ni = 0; ni < 4; ++ni) {
      int col = n0 + 16 * ni + l15;
      float bv = bias ? bias[col] : 0.0f;
#pragma unroll
      for (int r = 0; r < 8; ++r) {
        int row = m0 + 16 * mi + r + 8 * hl;
        C[(size_t)row * ldc + col] = acc[mi][ni][r] + bv;
      }
    }
}

// ---------------------------------------------------------------------------
// float -> bf16 conversion
// ---------------------------------------------------------------------------
__global__ void cvt_f32_bf16(const float* __restrict__ src,
                             __bf16* __restrict__ dst, size_t n) {
  size_t i = (size_t)blockIdx.x * blockDim.x + threadIdx.x;
  if (i < n) dst[i] = (__bf16)src[i];
}

__global__ void zero_state(float* h, __bf16* hbf, int n) {
  int i = blockIdx.x * blockDim.x + threadIdx.x;
  if (i < n) { h[i] = 0.0f; hbf[i] = (__bf16)0.0f; }
}

// ---------------------------------------------------------------------------
// Fused GRU gate update for both directions at step t.
// xg_* : [B*T][1536] precomputed input projections (r,z,n)
// hg   : [2][B][1536] recurrent projections for this step
// h/hbf: [2][B][512] state; hs: [B][T][1024] concat(fwd,bwd) output
// ---------------------------------------------------------------------------
__global__ void gru_gate(const float* __restrict__ xg_f,
                         const float* __restrict__ xg_b,
                         const float* __restrict__ hg,
                         float* __restrict__ h, __bf16* __restrict__ hbf,
                         float* __restrict__ hs, int t) {
  int idx = blockIdx.x * blockDim.x + threadIdx.x;
  if (idx >= 2 * Bb * Hh) return;
  int j   = idx & (Hh - 1);
  int b   = (idx >> 9) & (Bb - 1);
  int dir = idx >> 16;
  int time = dir ? (Tt - 1 - t) : t;

  const float* xg = dir ? xg_b : xg_f;
  size_t xrow = ((size_t)b * Tt + time) * G3;
  const float* hgp = hg + ((size_t)dir * Bb + b) * G3;

  float xr = xg[xrow + j], xz = xg[xrow + Hh + j], xn = xg[xrow + 2 * Hh + j];
  float hr = hgp[j],       hz = hgp[Hh + j],       hn = hgp[2 * Hh + j];

  float r = 1.0f / (1.0f + __expf(-(xr + hr)));
  float z = 1.0f / (1.0f + __expf(-(xz + hz)));
  float n = tanhf(xn + r * hn);

  size_t hidx = ((size_t)dir * Bb + b) * Hh + j;
  float hnew = (1.0f - z) * n + z * h[hidx];
  h[hidx]   = hnew;
  hbf[hidx] = (__bf16)hnew;
  hs[((size_t)b * Tt + time) * ROUT + dir * Hh + j] = hnew;
}

// ---------------------------------------------------------------------------
// gru_out = sum_t hs*mask / len -> features[:,0:1024]; also masked_bf16 for conv
// ---------------------------------------------------------------------------
__global__ void mask_mean(const float* __restrict__ hs,
                          const float* __restrict__ mask,
                          const int* __restrict__ lengths,
                          float* __restrict__ features,
                          __bf16* __restrict__ masked) {
  int b = blockIdx.x;
  float inv = 1.0f / (float)lengths[b];
  for (int c = threadIdx.x; c < ROUT; c += blockDim.x) {
    float s = 0.0f;
    for (int t = 0; t < Tt; ++t) {
      float m = mask[b * Tt + t];
      float mv = hs[((size_t)b * Tt + t) * ROUT + c] * m;
      masked[((size_t)b * Tt + t) * ROUT + c] = (__bf16)mv;
      s += mv;
    }
    features[(size_t)b * FIN + c] = s * inv;
  }
}

// ---------------------------------------------------------------------------
// Conv branch as WMMA GEMM: y[p][k] = sum_i sum_c x[b, p-(ws-1)+i, c]*w[k,i,c]
// then relu(y+bias), max over valid p (p < T+ws-1). One wave per (b, widx,
// 16-channel tile); M = 80 positions (5 tiles, invalid rows masked to -inf).
// grid = (4, B, 3), block = 256.
// ---------------------------------------------------------------------------
__global__ void conv_branch_wmma(const __bf16* __restrict__ masked,
                                 const __bf16* __restrict__ convw,
                                 const float* __restrict__ cb2,
                                 const float* __restrict__ cb3,
                                 const float* __restrict__ cb4,
                                 float* __restrict__ features) {
  const int widx = blockIdx.z;
  const int ws   = widx + 2;
  const int b    = blockIdx.y;
  const int lane = threadIdx.x & 31;
  const int wid  = threadIdx.x >> 5;
  const int hl   = (lane >> 4) & 1;
  const int l15  = lane & 15;
  const int n0   = (blockIdx.x * 8 + wid) * 16;

  const size_t woff = (widx == 0) ? 0
                    : (widx == 1) ? (size_t)Kc * 2 * ROUT
                                  : (size_t)Kc * 5 * ROUT;
  const __bf16* W = convw + woff;
  const float* cbias = (widx == 0) ? cb2 : (widx == 1) ? cb3 : cb4;
  const int P = Tt + ws - 1;

  v8f acc[5];
#pragma unroll
  for (int mt = 0; mt < 5; ++mt) acc[mt] = zero_v8f();
  const v16bf zf = zero_v16bf();

  for (int i = 0; i < ws; ++i) {
    for (int c0 = 0; c0 < ROUT; c0 += 32) {
      const __bf16* pb = W + ((size_t)(n0 + l15) * ws + i) * ROUT + c0;
      v16bf bfr = load_b_frag(pb, hl);
#pragma unroll
      for (int mt = 0; mt < 5; ++mt) {
        int trow = mt * 16 + l15 - (ws - 1) + i;
        v16bf af;
        if (trow < 0 || trow >= Tt) {
          af = zf;
        } else {
          const __bf16* pa = masked + ((size_t)b * Tt + trow) * ROUT + c0;
          af = load_a_frag(pa, hl);
        }
        acc[mt] = wmma_bf16(af, bfr, acc[mt]);
      }
    }
  }

  float bv = cbias[n0 + l15];
  float vmax = -INFINITY;
#pragma unroll
  for (int mt = 0; mt < 5; ++mt)
#pragma unroll
    for (int r = 0; r < 8; ++r) {
      int p = mt * 16 + r + 8 * hl;
      if (p < P) {
        float v = acc[mt][r] + bv;
        v = v < 0.0f ? 0.0f : v;
        vmax = fmaxf(vmax, v);
      }
    }
  vmax = fmaxf(vmax, __shfl_xor(vmax, 16, 32));
  if (hl == 0)
    features[(size_t)b * FIN + ROUT + widx * Kc + n0 + l15] = vmax;
}

// ---------------------------------------------------------------------------
// features[:, 2560:4608] = videos_origin ; features[:, 4608:5632] = motions
// ---------------------------------------------------------------------------
__global__ void fill_feat(const float* __restrict__ origin,
                          const float* __restrict__ motions,
                          float* __restrict__ features) {
  int idx = blockIdx.x * blockDim.x + threadIdx.x;
  if (idx >= Bb * 3072) return;
  int b = idx / 3072, c = idx % 3072;
  float v = (c < 2048) ? origin[(size_t)b * 2048 + c]
                       : motions[(size_t)b * 1024 + (c - 2048)];
  features[(size_t)b * FIN + 2560 + c] = v;
}

// ---------------------------------------------------------------------------
// BatchNorm (inference) + L2 row-normalize -> d_out
// ---------------------------------------------------------------------------
__global__ void bn_l2(const float* __restrict__ x, const float* __restrict__ gamma,
                      const float* __restrict__ beta, const float* __restrict__ mean,
                      const float* __restrict__ var, float* __restrict__ out) {
  int b = blockIdx.x, tid = threadIdx.x;
  __shared__ float red[256];
  float vals[8];
  float ss = 0.0f;
#pragma unroll
  for (int q = 0; q < 8; ++q) {
    int c = tid + q * 256;
    float y = gamma[c] * (x[(size_t)b * FOUT + c] - mean[c]) *
                  rsqrtf(var[c] + 1e-5f) + beta[c];
    vals[q] = y;
    ss += y * y;
  }
  red[tid] = ss;
  __syncthreads();
  for (int s = 128; s > 0; s >>= 1) {
    if (tid < s) red[tid] += red[tid + s];
    __syncthreads();
  }
  float rn = rsqrtf(red[0]);
#pragma unroll
  for (int q = 0; q < 8; ++q)
    out[(size_t)b * FOUT + tid + q * 256] = vals[q] * rn;
}

// ---------------------------------------------------------------------------
extern "C" void kernel_launch(void* const* d_in, const int* in_sizes, int n_in,
                              void* d_out, int out_size, void* d_ws, size_t ws_size,
                              hipStream_t stream) {
  (void)in_sizes; (void)n_in; (void)out_size; (void)ws_size;
  const float* videos   = (const float*)d_in[0];
  const float* motions  = (const float*)d_in[1];
  const float* origin   = (const float*)d_in[2];
  const int*   lengths  = (const int*)d_in[3];
  const float* Wih_f    = (const float*)d_in[4];
  const float* Whh_f    = (const float*)d_in[5];
  const float* bih_f    = (const float*)d_in[6];
  const float* bhh_f    = (const float*)d_in[7];
  const float* Wih_b    = (const float*)d_in[8];
  const float* Whh_b    = (const float*)d_in[9];
  const float* bih_b    = (const float*)d_in[10];
  const float* bhh_b    = (const float*)d_in[11];
  const float* fc_w     = (const float*)d_in[12];
  const float* fc_b     = (const float*)d_in[13];
  const float* bn_gamma = (const float*)d_in[14];
  const float* bn_beta  = (const float*)d_in[15];
  const float* bn_mean  = (const float*)d_in[16];
  const float* bn_var   = (const float*)d_in[17];
  const float* conv_w2  = (const float*)d_in[18];
  const float* conv_b2  = (const float*)d_in[19];
  const float* conv_w3  = (const float*)d_in[20];
  const float* conv_b3  = (const float*)d_in[21];
  const float* conv_w4  = (const float*)d_in[22];
  const float* conv_b4  = (const float*)d_in[23];
  const float* vmask    = (const float*)d_in[24];

  // ---- workspace layout (running byte offset, 256-aligned chunks) ----
  char* ws = (char*)d_ws;
  size_t off = 0;
  auto alloc = [&](size_t bytes) -> char* {
    char* p = ws + off;
    off += (bytes + 255) & ~(size_t)255;
    return p;
  };
  __bf16* videos_bf = (__bf16*)alloc((size_t)Bb * Tt * Dd * 2);
  __bf16* wih_bf    = (__bf16*)alloc((size_t)2 * G3 * Dd * 2);
  __bf16* whh_bf    = (__bf16*)alloc((size_t)2 * G3 * Hh * 2);
  float*  xg_f      = (float*)alloc((size_t)Bb * Tt * G3 * 4);
  float*  xg_b      = (float*)alloc((size_t)Bb * Tt * G3 * 4);
  float*  hg        = (float*)alloc((size_t)2 * Bb * G3 * 4);
  float*  hstate    = (float*)alloc((size_t)2 * Bb * Hh * 4);
  __bf16* hbf       = (__bf16*)alloc((size_t)2 * Bb * Hh * 2);
  float*  hs        = (float*)alloc((size_t)Bb * Tt * ROUT * 4);
  __bf16* masked    = (__bf16*)alloc((size_t)Bb * Tt * ROUT * 2);
  __bf16* convw_bf  = (__bf16*)alloc((size_t)Kc * 9 * ROUT * 2);
  float*  features  = (float*)alloc((size_t)Bb * FIN * 4);
  __bf16* feat_bf   = (__bf16*)alloc((size_t)Bb * FIN * 2);
  __bf16* fcw_bf    = (__bf16*)alloc((size_t)FOUT * FIN * 2);
  float*  fcout     = (float*)alloc((size_t)Bb * FOUT * 4);

  const dim3 blk(256);
  auto cvt = [&](const float* s, __bf16* d, size_t n) {
    cvt_f32_bf16<<<dim3((unsigned)((n + 255) / 256)), blk, 0, stream>>>(s, d, n);
  };

  // ---- bf16 conversions ----
  cvt(videos, videos_bf, (size_t)Bb * Tt * Dd);
  cvt(Wih_f, wih_bf,                    (size_t)G3 * Dd);
  cvt(Wih_b, wih_bf + (size_t)G3 * Dd,  (size_t)G3 * Dd);
  cvt(Whh_f, whh_bf,                    (size_t)G3 * Hh);
  cvt(Whh_b, whh_bf + (size_t)G3 * Hh,  (size_t)G3 * Hh);
  cvt(conv_w2, convw_bf,                          (size_t)Kc * 2 * ROUT);
  cvt(conv_w3, convw_bf + (size_t)Kc * 2 * ROUT,  (size_t)Kc * 3 * ROUT);
  cvt(conv_w4, convw_bf + (size_t)Kc * 5 * ROUT,  (size_t)Kc * 4 * ROUT);
  cvt(fc_w, fcw_bf, (size_t)FOUT * FIN);

  // ---- input projections: xg = videos @ Wih^T + bih  (M=8192,N=1536,K=2048) ----
  {
    dim3 grid(G3 / 256, (Bb * Tt) / 64);
    gemm_bf16_wmma<<<grid, blk, 0, stream>>>(videos_bf, Dd, wih_bf, Dd,
                                             bih_f, xg_f, G3, Dd);
    gemm_bf16_wmma<<<grid, blk, 0, stream>>>(videos_bf, Dd,
                                             wih_bf + (size_t)G3 * Dd, Dd,
                                             bih_b, xg_b, G3, Dd);
  }

  // ---- GRU recurrence ----
  zero_state<<<dim3((2 * Bb * Hh + 255) / 256), blk, 0, stream>>>(
      hstate, hbf, 2 * Bb * Hh);
  {
    dim3 ggrid(G3 / 256, Bb / 64);
    for (int t = 0; t < Tt; ++t) {
      // hg = h @ Whh^T + bhh  per direction  (M=128, N=1536, K=512)
      gemm_bf16_wmma<<<ggrid, blk, 0, stream>>>(
          hbf, Hh, whh_bf, Hh, bhh_f, hg, G3, Hh);
      gemm_bf16_wmma<<<ggrid, blk, 0, stream>>>(
          hbf + (size_t)Bb * Hh, Hh, whh_bf + (size_t)G3 * Hh, Hh,
          bhh_b, hg + (size_t)Bb * G3, G3, Hh);
      gru_gate<<<dim3((2 * Bb * Hh + 255) / 256), blk, 0, stream>>>(
          xg_f, xg_b, hg, hstate, hbf, hs, t);
    }
  }

  // ---- masked mean -> features[:,0:1024]; masked bf16 tensor for convs ----
  mask_mean<<<dim3(Bb), blk, 0, stream>>>(hs, vmask, lengths, features, masked);

  // ---- conv branches (WMMA) -> features[:,1024:2560] ----
  conv_branch_wmma<<<dim3(4, Bb, 3), blk, 0, stream>>>(
      masked, convw_bf, conv_b2, conv_b3, conv_b4, features);

  // ---- copy videos_origin / motions into features ----
  fill_feat<<<dim3((Bb * 3072 + 255) / 256), blk, 0, stream>>>(
      origin, motions, features);

  // ---- FC: fcout = features @ fc_w^T + fc_b  (M=128, N=2048, K=5632) ----
  cvt(features, feat_bf, (size_t)Bb * FIN);
  {
    dim3 grid(FOUT / 256, Bb / 64);
    gemm_bf16_wmma<<<grid, blk, 0, stream>>>(feat_bf, FIN, fcw_bf, FIN,
                                             fc_b, fcout, FOUT, FIN);
  }

  // ---- BN + L2 normalize -> d_out ----
  bn_l2<<<dim3(Bb), blk, 0, stream>>>(fcout, bn_gamma, bn_beta, bn_mean,
                                      bn_var, (float*)d_out);
}
